// PatchedGptOssTopKRouter_30777735643925
// MI455X (gfx1250) — matrix operand order repaired and checked by reading, not compiled, not run on hardware
//
#include <hip/hip_runtime.h>
#include <hip/hip_bf16.h>
#include <math.h>

typedef __attribute__((ext_vector_type(16))) __bf16 v16bf;
typedef __attribute__((ext_vector_type(8)))  float  v8f;
typedef __attribute__((ext_vector_type(4)))  float  f4;

constexpr int kT = 16384;      // tokens
constexpr int kH = 2880;       // hidden
constexpr int kE = 128;        // experts
constexpr int kTopK = 4;
constexpr int kWaves = 4;                      // waves per block (wave32)
constexpr int kRowsPerWave = 16;
constexpr int kRowsPerBlock = kWaves * kRowsPerWave;   // 64
constexpr int kPitch = kE + 4;                 // LDS row pitch (bank-conflict-free)

__global__ __launch_bounds__(kWaves * 32)
void router_topk_kernel(const float* __restrict__ x, const float* __restrict__ w,
                        const float* __restrict__ bias, float* __restrict__ scores,
                        int* __restrict__ topIdx)
{
    __shared__ float lds[kWaves * kRowsPerWave * kPitch];

    const int tid  = threadIdx.x;
    const int wave = tid >> 5;
    const int lane = tid & 31;
    const int half = lane >> 4;      // 0: lanes 0-15, 1: lanes 16-31
    const int ln   = lane & 15;
    const int rowBase = blockIdx.x * kRowsPerBlock + wave * kRowsPerWave;
    const float* aRow = x + (size_t)(rowBase + ln) * kH;

    // one-shot warm-up prefetch of this row's first chunk (global_prefetch path,
    // zero cost inside the hot loop)
    __builtin_prefetch(aRow, 0, 0);

    v8f acc[8];
#pragma unroll
    for (int n = 0; n < 8; ++n) acc[n] = (v8f)0.0f;

    for (int k0 = 0; k0 < kH; k0 += 32) {
        // ---- A tile: 16 rows x 32 K of x, fp32 -> bf16, WMMA A layout ----
        // lanes 0-15: K = k0..k0+15; lanes 16-31: K = k0+16..k0+31
        v16bf a;
        {
            const float4* p0 = reinterpret_cast<const float4*>(aRow + k0 + half * 16);
            const float4* p1 = reinterpret_cast<const float4*>(aRow + k0 + half * 16 + 8);
            float4 f0 = p0[0], f1 = p0[1], f2 = p1[0], f3 = p1[1];
            a[0]=(__bf16)f0.x;  a[1]=(__bf16)f0.y;  a[2]=(__bf16)f0.z;  a[3]=(__bf16)f0.w;
            a[4]=(__bf16)f1.x;  a[5]=(__bf16)f1.y;  a[6]=(__bf16)f1.z;  a[7]=(__bf16)f1.w;
            a[8]=(__bf16)f2.x;  a[9]=(__bf16)f2.y;  a[10]=(__bf16)f2.z; a[11]=(__bf16)f2.w;
            a[12]=(__bf16)f3.x; a[13]=(__bf16)f3.y; a[14]=(__bf16)f3.z; a[15]=(__bf16)f3.w;
        }

        // ---- B tiles: 32 K x 16 experts each; 8 tiles cover all 128 experts ----
        // All 8 tiles get distinct live staging registers so the 8 trailing WMMAs
        // have no WAR hazard with the next iteration's cvts (no v_nop insertion).
        // lanes 0-15: N=ln, K=k0..k0+15; lanes 16-31: N=ln, K=k0+16..k0+31
        v16bf bm[8];
#pragma unroll
        for (int n = 0; n < 8; ++n) {
            const float4* wp = reinterpret_cast<const float4*>(
                w + (size_t)(n * 16 + ln) * kH + k0 + half * 16);
            float4 g0 = wp[0], g1 = wp[1], g2 = wp[2], g3 = wp[3];
            bm[n][0]=(__bf16)g0.x;  bm[n][1]=(__bf16)g0.y;  bm[n][2]=(__bf16)g0.z;  bm[n][3]=(__bf16)g0.w;
            bm[n][4]=(__bf16)g1.x;  bm[n][5]=(__bf16)g1.y;  bm[n][6]=(__bf16)g1.z;  bm[n][7]=(__bf16)g1.w;
            bm[n][8]=(__bf16)g2.x;  bm[n][9]=(__bf16)g2.y;  bm[n][10]=(__bf16)g2.z; bm[n][11]=(__bf16)g2.w;
            bm[n][12]=(__bf16)g3.x; bm[n][13]=(__bf16)g3.y; bm[n][14]=(__bf16)g3.z; bm[n][15]=(__bf16)g3.w;
        }
#pragma unroll
        for (int n = 0; n < 8; ++n) {
            acc[n] = __builtin_amdgcn_wmma_f32_16x16x32_bf16(
                false, a, false, bm[n], (short)0, acc[n], false, false);
        }
    }

    // ---- bias add + stash this wave's 16x128 logits tile in LDS ----
    // C/D layout: VGPR j, lanes 0-15 -> (M=j, N=lane); lanes 16-31 -> (M=j+8, N=lane-16)
    float* myLds = lds + wave * kRowsPerWave * kPitch;
#pragma unroll
    for (int n = 0; n < 8; ++n) {
        const float bv = bias[n * 16 + ln];
#pragma unroll
        for (int j = 0; j < 8; ++j) {
            myLds[(j + 8 * half) * kPitch + n * 16 + ln] = acc[n][j] + bv;
        }
    }

    // ---- per-row top-4 (stable: lowest index wins ties) + softmax ----
    // float4 LDS reads: 32 x ds_load_b128 instead of 128 scalar loads.
    // pitch 132 floats = 528B (16B aligned); lane ln hits bank group 4*ln -> conflict-free
    float p0 = 0.f, p1 = 0.f, p2 = 0.f, p3 = 0.f;
    int   i0 = 0, i1 = 0, i2 = 0, i3 = 0;
    if (half == 0) {
        const f4* rp = reinterpret_cast<const f4*>(myLds + ln * kPitch);
        float v0 = -3.402823466e38f, v1 = v0, v2 = v0, v3 = v0;
        for (int u = 0; u < kE / 4; ++u) {
            f4 q = rp[u];
#pragma unroll
            for (int j = 0; j < 4; ++j) {
                float v = q[j];
                int   e = 4 * u + j;
                if (v > v0)      { v3=v2; i3=i2; v2=v1; i2=i1; v1=v0; i1=i0; v0=v; i0=e; }
                else if (v > v1) { v3=v2; i3=i2; v2=v1; i2=i1; v1=v;  i1=e; }
                else if (v > v2) { v3=v2; i3=i2; v2=v;  i2=e; }
                else if (v > v3) { v3=v;  i3=e; }
            }
        }
        float e1 = __expf(v1 - v0), e2 = __expf(v2 - v0), e3 = __expf(v3 - v0);
        float inv = 1.0f / (1.0f + e1 + e2 + e3);
        p0 = inv; p1 = e1 * inv; p2 = e2 * inv; p3 = e3 * inv;
    }

    // ---- coalesced zero-fill of this wave's 16x128 score tile (non-temporal:
    // write-only stream, keep it out of WGP$/L2 working set) ----
    f4* outTile = reinterpret_cast<f4*>(scores + (size_t)rowBase * kE);
    const f4 z = (f4)0.0f;
#pragma unroll
    for (int i = 0; i < (kRowsPerWave * kE / 4) / 32; ++i) {
        __builtin_nontemporal_store(z, outTile + lane + 32 * i);
    }

    // ---- scatter softmaxed top-4 + indices (same wave -> stores stay ordered) ----
    if (half == 0) {
        const int grow = rowBase + ln;
        float* srow = scores + (size_t)grow * kE;
        __builtin_nontemporal_store(p0, srow + i0);
        __builtin_nontemporal_store(p1, srow + i1);
        __builtin_nontemporal_store(p2, srow + i2);
        __builtin_nontemporal_store(p3, srow + i3);
        int* irow = topIdx + grow * kTopK;
        __builtin_nontemporal_store(i0, irow + 0);
        __builtin_nontemporal_store(i1, irow + 1);
        __builtin_nontemporal_store(i2, irow + 2);
        __builtin_nontemporal_store(i3, irow + 3);
    }
}

extern "C" void kernel_launch(void* const* d_in, const int* in_sizes, int n_in,
                              void* d_out, int out_size, void* d_ws, size_t ws_size,
                              hipStream_t stream) {
    const float* x    = (const float*)d_in[0];   // [T, H] fp32
    const float* w    = (const float*)d_in[1];   // [E, H] fp32
    const float* bias = (const float*)d_in[2];   // [E]    fp32
    // d_in[3] = top_k (== 4, compiled in)

    float* scores = (float*)d_out;                           // [T, E] fp32
    int*   topIdx = (int*)(scores + (size_t)kT * kE);        // [T, 4] int32 (raw bits)

    dim3 grid(kT / kRowsPerBlock);   // 256 blocks
    dim3 block(kWaves * 32);         // 128 threads = 4 waves
    hipLaunchKernelGGL(router_topk_kernel, grid, block, 0, stream,
                       x, w, bias, scores, topIdx);
}
